// BAKT_34686155883025
// MI455X (gfx1250) — compile-verified, matrix operand order checked by reading, not run on hardware
//
#include <hip/hip_runtime.h>
#include <math.h>

// ---------------- problem constants ----------------
constexpr int BB  = 8;
constexpr int SS  = 1024;
constexpr int DD  = 512;
constexpr int HH  = 8;
constexpr int FFN = 2048;
constexpr int LL  = 4;
constexpr int DKH = DD / HH;          // 64
constexpr int BS  = BB * SS;          // 8192 token rows
constexpr float EPSL = 1e-5f;

typedef __attribute__((ext_vector_type(16))) _Float16 v16h;
typedef __attribute__((ext_vector_type(8)))  _Float16 v8h;
typedef __attribute__((ext_vector_type(8)))  float    v8f;

// Load a 16-lane-half-split f16 fragment (CDNA5 16-bit A/B-matrix layout):
// lane<16 -> k = {k0+0..7, k0+16..23}; lane>=16 -> k = {k0+8..15, k0+24..31}.
// rowptr points at the lane's assigned row; K is contiguous in memory.
__device__ __forceinline__ v16h load_frag16(const _Float16* rowptr, int k0, int kb) {
  v8h lo = *(const v8h*)(rowptr + k0 + kb);
  v8h hi = *(const v8h*)(rowptr + k0 + kb + 16);
  v16h f;
#pragma unroll
  for (int i = 0; i < 8; ++i) { f[i] = lo[i]; f[i + 8] = hi[i]; }
  return f;
}

__device__ __forceinline__ v8f wmma_f16(v16h a, v16h b, v8f c) {
  return __builtin_amdgcn_wmma_f32_16x16x32_f16(false, a, false, b, (short)0, c, false, false);
}

__device__ __forceinline__ float half_max(float v) {
  v = fmaxf(v, __shfl_xor(v, 1, 32));
  v = fmaxf(v, __shfl_xor(v, 2, 32));
  v = fmaxf(v, __shfl_xor(v, 4, 32));
  v = fmaxf(v, __shfl_xor(v, 8, 32));
  return v;   // reduced over the 16-lane half (matches C-layout rows)
}
__device__ __forceinline__ float half_sum(float v) {
  v += __shfl_xor(v, 1, 32);
  v += __shfl_xor(v, 2, 32);
  v += __shfl_xor(v, 4, 32);
  v += __shfl_xor(v, 8, 32);
  return v;
}

// ---------------- f32 -> f16 convert ----------------
__global__ void cvt_f16_kernel(const float* __restrict__ in, _Float16* __restrict__ out, size_t n) {
  size_t i = (size_t)blockIdx.x * blockDim.x + threadIdx.x;
  size_t stride = (size_t)gridDim.x * blockDim.x;
  for (; i < n; i += stride) out[i] = (_Float16)in[i];
}

// ---------------- x = q+pe ; y16 = f16(qa+pe) ----------------
__global__ void addpe_kernel(const float* __restrict__ q, const float* __restrict__ qa,
                             const float* __restrict__ pe,
                             float* __restrict__ X, _Float16* __restrict__ X16,
                             _Float16* __restrict__ Y16) {
  size_t n = (size_t)BS * DD;
  size_t i = (size_t)blockIdx.x * blockDim.x + threadIdx.x;
  size_t stride = (size_t)gridDim.x * blockDim.x;
  for (; i < n; i += stride) {
    float p  = pe[i % ((size_t)SS * DD)];
    float xv = q[i] + p;
    X[i]   = xv;
    X16[i] = (_Float16)xv;
    Y16[i] = (_Float16)(qa[i] + p);
  }
}

// ---------------- generic WMMA GEMM: C[M,N] = A[M,K] * W[N,K]^T + bias ----------------
// 8 waves / 256-thread WG; each wave register-tiles a 32x32 output (2x2 WMMA tiles),
// so the k-loop issues 4 WMMAs per 8 b128 fragment loads (2 loads / WMMA).
// Output modes: 0 = f32 row-major, 1 = f16 row-major, 2 = f16 per-head transposed
// (Vt[b,h,d,s], sequence-contiguous — feeds attention P*V B-fragments directly).
__global__ void gemm_wmma_kernel(const _Float16* __restrict__ A, const _Float16* __restrict__ W,
                                 const float* __restrict__ bias,
                                 _Float16* __restrict__ out16, float* __restrict__ out32,
                                 int M, int N, int K, int relu, int mode) {
  int wave = threadIdx.x >> 5;
  int lane = threadIdx.x & 31;
  int r16  = lane & 15;
  int kb   = (lane < 16) ? 0 : 8;
  int m0 = blockIdx.x * 128 + (wave >> 1) * 32;   // 4 wave-rows
  int n0 = blockIdx.y * 64  + (wave &  1) * 32;   // 2 wave-cols

  const _Float16* a0p = A + (size_t)(m0 + r16)      * K;
  const _Float16* a1p = A + (size_t)(m0 + 16 + r16) * K;
  const _Float16* w0p = W + (size_t)(n0 + r16)      * K;
  const _Float16* w1p = W + (size_t)(n0 + 16 + r16) * K;

  v8f c00 = {}, c01 = {}, c10 = {}, c11 = {};
  for (int k0 = 0; k0 < K; k0 += 32) {
    v16h a0 = load_frag16(a0p, k0, kb);
    v16h a1 = load_frag16(a1p, k0, kb);
    v16h b0 = load_frag16(w0p, k0, kb);
    v16h b1 = load_frag16(w1p, k0, kb);
    c00 = wmma_f16(a0, b0, c00);
    c01 = wmma_f16(a0, b1, c01);
    c10 = wmma_f16(a1, b0, c10);
    c11 = wmma_f16(a1, b1, c11);
  }

  int   col0 = n0 + r16, col1 = n0 + 16 + r16;
  float bn0  = bias[col0], bn1 = bias[col1];
  int   mb0  = m0 + ((lane < 16) ? 0 : 8);
  int   mb1  = mb0 + 16;

  v8f  cs[4]   = {c00, c01, c10, c11};
  int  mbs[4]  = {mb0, mb0, mb1, mb1};
  int  cols[4] = {col0, col1, col0, col1};
  float bns[4] = {bn0, bn1, bn0, bn1};
#pragma unroll
  for (int t = 0; t < 4; ++t) {
#pragma unroll
    for (int r = 0; r < 8; ++r) {
      float v = cs[t][r] + bns[t];
      if (relu) v = fmaxf(v, 0.0f);
      int m = mbs[t] + r, col = cols[t];
      if (mode == 0) {
        out32[(size_t)m * N + col] = v;
      } else if (mode == 1) {
        out16[(size_t)m * N + col] = (_Float16)v;
      } else {   // per-head transpose: [b,h,d,s]
        int b = m >> 10, s = m & (SS - 1);
        int h = col >> 6, d = col & (DKH - 1);
        out16[(((size_t)(b * HH + h)) * DKH + d) * SS + s] = (_Float16)v;
      }
    }
  }
}

// ---------------- flash attention (causal strict j<i, kq used as both Q and K) ----------------
// 8 independent waves per block; each wave handles one (b,h,16-query) tile.
// KQ is row-major [b,s,h*64+d]; Vt is per-head transposed [b,h,d,s] so the
// P*V B-fragments are two contiguous b128 loads, same as Q/K fragments.
__global__ void attn_wmma_kernel(const _Float16* __restrict__ KQ, const _Float16* __restrict__ Vt,
                                 _Float16* __restrict__ O) {
  __shared__ _Float16 pld[8][16 * 32];   // per-wave P transpose staging

  int wave  = threadIdx.x >> 5;
  int lane  = threadIdx.x & 31;
  int r16   = lane & 15;
  int half8 = (lane < 16) ? 0 : 8;
  int kb    = (lane < 16) ? 0 : 8;

  int tile = blockIdx.x * 8 + wave;      // 0 .. B*H*(S/16)-1
  int qt = tile & (SS / 16 - 1);
  int bh = tile >> 6;                    // S/16 == 64
  int h  = bh & (HH - 1);
  int b  = bh >> 3;                      // HH == 8
  int i0 = qt * 16;

  const float scale = 0.125f;            // 1/sqrt(64)
  const float NEGI  = -1.0e30f;

  const _Float16* Qhead = KQ + (size_t)b * SS * DD + h * DKH;
  const _Float16* Vth   = Vt + ((size_t)(b * HH + h)) * DKH * SS;
  _Float16*       Ohead = O  + (size_t)b * SS * DD + h * DKH;

  const _Float16* qrow = Qhead + (size_t)(i0 + r16) * DD;
  v16h aq0 = load_frag16(qrow, 0,  kb);
  v16h aq1 = load_frag16(qrow, 32, kb);

  float mrow[8], srow[8];
#pragma unroll
  for (int r = 0; r < 8; ++r) { mrow[r] = NEGI; srow[r] = 0.0f; }
  v8f o[4] = {};

  _Float16* pw = pld[wave];

  for (int j0 = 0; j0 < i0 + 16; j0 += 32) {   // key block of 32, uniform per wave
    const _Float16* krow0 = Qhead + (size_t)(j0 + r16) * DD;
    const _Float16* krow1 = Qhead + (size_t)(j0 + 16 + r16) * DD;
    v16h bk00 = load_frag16(krow0, 0,  kb);
    v16h bk01 = load_frag16(krow0, 32, kb);
    v16h bk10 = load_frag16(krow1, 0,  kb);
    v16h bk11 = load_frag16(krow1, 32, kb);

    v8f s0 = {}, s1 = {};
    s0 = wmma_f16(aq0, bk00, s0);
    s0 = wmma_f16(aq1, bk01, s0);
    s1 = wmma_f16(aq0, bk10, s1);
    s1 = wmma_f16(aq1, bk11, s1);

    float p0[8], p1[8], tmax[8];
#pragma unroll
    for (int r = 0; r < 8; ++r) {
      int m  = i0 + r + half8;
      float a0 = (j0 + r16      < m) ? s0[r] * scale : NEGI;
      float a1 = (j0 + 16 + r16 < m) ? s1[r] * scale : NEGI;
      p0[r] = a0; p1[r] = a1;
      tmax[r] = half_max(fmaxf(a0, a1));
    }

    float nmax[8], corr[8], tsum[8];
#pragma unroll
    for (int r = 0; r < 8; ++r) {
      nmax[r] = fmaxf(mrow[r], tmax[r]);
      corr[r] = __expf(mrow[r] - nmax[r]);
      float e0 = (p0[r] < -1.0e29f) ? 0.0f : __expf(p0[r] - nmax[r]);
      float e1 = (p1[r] < -1.0e29f) ? 0.0f : __expf(p1[r] - nmax[r]);
      p0[r] = e0; p1[r] = e1;
      tsum[r] = half_sum(e0 + e1);
    }

    // transpose P (D-layout: lane=col) into A-layout via LDS
#pragma unroll
    for (int r = 0; r < 8; ++r) {
      int m = r + half8;
      pw[m * 32 + r16]      = (_Float16)p0[r];
      pw[m * 32 + 16 + r16] = (_Float16)p1[r];
    }
    asm volatile("" ::: "memory");   // DS ops are in-order within a wave
    v16h ap = load_frag16(pw + r16 * 32, 0, kb);

    // rescale running output, then accumulate P(16x32) x V(32x64)
#pragma unroll
    for (int t = 0; t < 4; ++t)
#pragma unroll
      for (int r = 0; r < 8; ++r) o[t][r] *= corr[r];

#pragma unroll
    for (int t = 0; t < 4; ++t) {
      const _Float16* vcol = Vth + (size_t)(t * 16 + r16) * SS;  // seq-contiguous
      v16h vf = load_frag16(vcol, j0, kb);
      o[t] = wmma_f16(ap, vf, o[t]);
    }

#pragma unroll
    for (int r = 0; r < 8; ++r) {
      srow[r] = srow[r] * corr[r] + tsum[r];
      mrow[r] = nmax[r];
    }
  }

  // normalize; srow==0 only for global query row 0 -> reproduces reference zero_pad
#pragma unroll
  for (int t = 0; t < 4; ++t)
#pragma unroll
    for (int r = 0; r < 8; ++r) {
      float val = (srow[r] > 0.0f) ? (o[t][r] / srow[r]) : 0.0f;
      Ohead[(size_t)(i0 + r + half8) * DD + t * 16 + r16] = (_Float16)val;
    }
}

// ---------------- residual + LayerNorm (one 256-thread block per row) ----------------
__global__ void addln_kernel(const float* __restrict__ x, const float* __restrict__ t,
                             const float* __restrict__ g, const float* __restrict__ bta,
                             float* __restrict__ xo, _Float16* __restrict__ xo16) {
  __shared__ float red[256];
  int row = blockIdx.x, tid = threadIdx.x;
  const float* xr = x + (size_t)row * DD;
  const float* tr = t + (size_t)row * DD;
  float a0 = xr[tid] + tr[tid];
  float a1 = xr[tid + 256] + tr[tid + 256];

  red[tid] = a0 + a1; __syncthreads();
  for (int d = 128; d > 0; d >>= 1) { if (tid < d) red[tid] += red[tid + d]; __syncthreads(); }
  float mean = red[0] * (1.0f / DD); __syncthreads();

  float d0 = a0 - mean, d1 = a1 - mean;
  red[tid] = d0 * d0 + d1 * d1; __syncthreads();
  for (int d = 128; d > 0; d >>= 1) { if (tid < d) red[tid] += red[tid + d]; __syncthreads(); }
  float rstd = rsqrtf(red[0] * (1.0f / DD) + EPSL);

  float o0 = d0 * rstd * g[tid]       + bta[tid];
  float o1 = d1 * rstd * g[tid + 256] + bta[tid + 256];
  size_t base = (size_t)row * DD;
  xo[base + tid] = o0;          xo[base + tid + 256] = o1;
  xo16[base + tid] = (_Float16)o0; xo16[base + tid + 256] = (_Float16)o1;
}

// ---------------- host orchestration ----------------
extern "C" void kernel_launch(void* const* d_in, const int* in_sizes, int n_in,
                              void* d_out, int out_size, void* d_ws, size_t ws_size,
                              hipStream_t stream) {
  const float* q_emb  = (const float*)d_in[0];
  const float* qa_emb = (const float*)d_in[1];
  const float* pe     = (const float*)d_in[2];
  const float* Wk     = (const float*)d_in[3];
  const float* bk     = (const float*)d_in[4];
  const float* Wv     = (const float*)d_in[5];
  const float* bv     = (const float*)d_in[6];
  const float* Wo     = (const float*)d_in[7];
  const float* bo     = (const float*)d_in[8];
  const float* ln1s   = (const float*)d_in[9];
  const float* ln1b   = (const float*)d_in[10];
  const float* W1     = (const float*)d_in[11];
  const float* b1     = (const float*)d_in[12];
  const float* W2     = (const float*)d_in[13];
  const float* b2     = (const float*)d_in[14];
  const float* ln2s   = (const float*)d_in[15];
  const float* ln2b   = (const float*)d_in[16];

  char* wsb = (char*)d_ws;
  const size_t MB = (size_t)1 << 20;
  float*    X     = (float*)   (wsb + 0 * MB);     // 16 MB  fp32 running activations
  float*    T     = (float*)   (wsb + 16 * MB);    // 16 MB  fp32 GEMM output temp
  _Float16* X16   = (_Float16*)(wsb + 32 * MB);    //  8 MB
  _Float16* Y16   = (_Float16*)(wsb + 40 * MB);    //  8 MB  (fixed V input)
  _Float16* KQ16  = (_Float16*)(wsb + 48 * MB);    //  8 MB
  _Float16* Vt16  = (_Float16*)(wsb + 56 * MB);    //  8 MB  per-head transposed V
  _Float16* AO16  = (_Float16*)(wsb + 64 * MB);    //  8 MB
  _Float16* FFH16 = (_Float16*)(wsb + 72 * MB);    // 32 MB
  _Float16* Wk16  = (_Float16*)(wsb + 104 * MB);   // 22 MB of f16 weights follow
  _Float16* Wv16  = Wk16 + (size_t)LL * DD * DD;
  _Float16* Wo16  = Wv16 + (size_t)LL * DD * DD;
  _Float16* W116  = Wo16 + (size_t)LL * DD * DD;
  _Float16* W216  = W116 + (size_t)LL * FFN * DD;

  // one-time weight conversion to f16
  cvt_f16_kernel<<<2048, 256, 0, stream>>>(Wk, Wk16, (size_t)LL * DD * DD);
  cvt_f16_kernel<<<2048, 256, 0, stream>>>(Wv, Wv16, (size_t)LL * DD * DD);
  cvt_f16_kernel<<<2048, 256, 0, stream>>>(Wo, Wo16, (size_t)LL * DD * DD);
  cvt_f16_kernel<<<2048, 256, 0, stream>>>(W1, W116, (size_t)LL * FFN * DD);
  cvt_f16_kernel<<<2048, 256, 0, stream>>>(W2, W216, (size_t)LL * DD * FFN);

  addpe_kernel<<<2048, 256, 0, stream>>>(q_emb, qa_emb, pe, X, X16, Y16);

  dim3 gD (BS / 128, DD  / 64);   // projections: M=8192, N=512  -> (64, 8)
  dim3 gFF(BS / 128, FFN / 64);   // FFN up:      M=8192, N=2048 -> (64, 32)
  int attnBlocks = BB * HH * (SS / 16) / 8;   // 512 blocks x 8 waves

  for (int l = 0; l < LL; ++l) {
    const size_t oDD = (size_t)l * DD * DD;
    const size_t oFD = (size_t)l * FFN * DD;

    // kq = f16(x @ Wk^T + bk)   (q and k identical: kq_same)
    gemm_wmma_kernel<<<gD, 256, 0, stream>>>(X16, Wk16 + oDD, bk + l * DD,
                                             KQ16, nullptr, BS, DD, DD, 0, 1);
    // v = f16(y @ Wv^T + bv), written per-head transposed [b,h,d,s]
    gemm_wmma_kernel<<<gD, 256, 0, stream>>>(Y16, Wv16 + oDD, bv + l * DD,
                                             Vt16, nullptr, BS, DD, DD, 0, 2);
    // attention -> AO16
    attn_wmma_kernel<<<attnBlocks, 256, 0, stream>>>(KQ16, Vt16, AO16);
    // out = AO @ Wo^T + bo (fp32 for residual)
    gemm_wmma_kernel<<<gD, 256, 0, stream>>>(AO16, Wo16 + oDD, bo + l * DD,
                                             nullptr, T, BS, DD, DD, 0, 0);
    // x = LN(x + out)
    addln_kernel<<<BS, 256, 0, stream>>>(X, T, ln1s + l * DD, ln1b + l * DD, X, X16);
    // ffh = f16(relu(x @ W1^T + b1))
    gemm_wmma_kernel<<<gFF, 256, 0, stream>>>(X16, W116 + oFD, b1 + l * FFN,
                                              FFH16, nullptr, BS, FFN, DD, 1, 1);
    // ff = ffh @ W2^T + b2 (fp32)
    gemm_wmma_kernel<<<gD, 256, 0, stream>>>(FFH16, W216 + oFD, b2 + l * DD,
                                             nullptr, T, BS, DD, FFN, 0, 0);
    // x = LN(x + ff); last layer writes final output
    float* xdst = (l == LL - 1) ? (float*)d_out : X;
    addln_kernel<<<BS, 256, 0, stream>>>(X, T, ln2s + l * DD, ln2b + l * DD, xdst, X16);
  }
}
